// GlobalPointer_64493228917443
// MI455X (gfx1250) — compile-verified, matrix operand order checked by reading, not run on hardware
//
#include <hip/hip_runtime.h>
#include <hip/hip_bf16.h>
#include <cmath>

// ---------------------------------------------------------------------------
// Problem constants (from reference): B=64, S=256, H=768, T=13, D=64
// ---------------------------------------------------------------------------
#define BB   64
#define SS   256
#define HH   768
#define TT   13
#define DD   64
#define NN   (TT * 2 * DD)        // 1664 columns of dense projection
#define MM   (BB * SS)            // 16384 rows
#define NEGC 1000000000000.0f

typedef __attribute__((ext_vector_type(16))) _Float16 v16h;
typedef __attribute__((ext_vector_type(8)))  _Float16 v8h;
typedef __attribute__((ext_vector_type(4)))  _Float16 v4h;
typedef __attribute__((ext_vector_type(8)))  float    v8f;

// ---------------------------------------------------------------------------
// Load one 16x32 f16 A-fragment (or B-fragment from a [n][k] row-major tile)
// using the CDNA5 WMMA VGPR swizzle:
//   lane 0-15 : row = lane,    halves 0..7 = K kb..kb+7,  8..15 = K kb+16..kb+23 (kb=0)
//   lane16-31 : row = lane-16, kb = 8
// Two 16B loads per lane; all source rows are 16B aligned.
// ---------------------------------------------------------------------------
__device__ __forceinline__ v16h load_frag(const _Float16* __restrict__ tile,
                                          int ld, int lane) {
  const int r  = lane & 15;
  const int kb = (lane >> 4) << 3;            // 0 or 8
  const _Float16* p = tile + (size_t)r * ld + kb;
  v8h lo = *reinterpret_cast<const v8h*>(p);
  v8h hi = *reinterpret_cast<const v8h*>(p + 16);
  v16h out;
#pragma unroll
  for (int i = 0; i < 8; ++i) { out[i] = lo[i]; out[i + 8] = hi[i]; }
  return out;
}

// ---------------------------------------------------------------------------
// Kernel 1: RoPE sin/cos table  (costab/sintab: [S][D] f32, repeat-2 layout)
// ---------------------------------------------------------------------------
__global__ void rope_table_kernel(float* __restrict__ costab,
                                  float* __restrict__ sintab) {
  int i = blockIdx.x * blockDim.x + threadIdx.x;          // S * D/2 = 8192
  if (i < SS * (DD / 2)) {
    int s = i >> 5;           // / (DD/2)... DD/2 = 32
    int f = i & 31;
    // inv = 10000^(-2f/D) = exp(-ln(10000) * 2f/D)
    float inv = __expf(-9.210340371976184f * (2.0f * (float)f / (float)DD));
    float ang = (float)s * inv;
    float c = cosf(ang), sn = sinf(ang);
    costab[s * DD + 2 * f]     = c;
    costab[s * DD + 2 * f + 1] = c;
    sintab[s * DD + 2 * f]     = sn;
    sintab[s * DD + 2 * f + 1] = sn;
  }
}

// ---------------------------------------------------------------------------
// Kernel 2: f32 hidden -> f16 (vectorized 4 at a time)
// ---------------------------------------------------------------------------
__global__ void cvt_f16_kernel(const float* __restrict__ x,
                               _Float16* __restrict__ y, int n4) {
  int i = blockIdx.x * blockDim.x + threadIdx.x;
  if (i < n4) {
    float4 v = reinterpret_cast<const float4*>(x)[i];
    v4h h;
    h[0] = (_Float16)v.x; h[1] = (_Float16)v.y;
    h[2] = (_Float16)v.z; h[3] = (_Float16)v.w;
    reinterpret_cast<v4h*>(y)[i] = h;
  }
}

// ---------------------------------------------------------------------------
// Kernel 3: dense_w [H][N] f32 -> wT [N][H] f16 (so B-fragments load row-major)
// ---------------------------------------------------------------------------
__global__ void transpose_w_kernel(const float* __restrict__ w,
                                   _Float16* __restrict__ wT) {
  int i = blockIdx.x * blockDim.x + threadIdx.x;          // N*H = 1277952
  if (i < NN * HH) {
    int n = i / HH, k = i - n * HH;
    wT[i] = (_Float16)w[(size_t)k * NN + n];
  }
}

// ---------------------------------------------------------------------------
// Kernel 4: projection GEMM (M=16384, N=1664, K=768) via v_wmma_f32_16x16x32_f16
// Each wave computes a 16x64 strip (A-fragment reused for 4 WMMAs).
// Epilogue: +bias, RoPE (partner via shfl_xor lane^1), store q/k f16 in
// [b*T+t][s][d] layout ready for GEMM2.
// ---------------------------------------------------------------------------
__global__ __launch_bounds__(256) void gemm1_rope_kernel(
    const _Float16* __restrict__ Ah,     // [M][H]
    const _Float16* __restrict__ wT,     // [N][H]
    const float*    __restrict__ bias,   // [N]
    const float*    __restrict__ costab, // [S][D]
    const float*    __restrict__ sintab, // [S][D]
    _Float16* __restrict__ qbuf,         // [B*T][S][D]
    _Float16* __restrict__ kbuf) {
  const int lane  = threadIdx.x & 31;
  const int wid   = blockIdx.x * 8 + (threadIdx.x >> 5);  // 26624 waves total
  const int strip = wid % 26;                             // 26 strips of 64 cols
  const int mtile = wid / 26;                             // 1024 row tiles
  const int n0    = strip * 64;

  v8f acc[4] = {};
  const _Float16* abase = Ah + (size_t)mtile * 16 * HH;
#pragma unroll 4
  for (int k0 = 0; k0 < HH; k0 += 32) {
    v16h a = load_frag(abase + k0, HH, lane);
#pragma unroll
    for (int j = 0; j < 4; ++j) {
      v16h b = load_frag(wT + (size_t)(n0 + j * 16) * HH + k0, HH, lane);
      acc[j] = __builtin_amdgcn_wmma_f32_16x16x32_f16(
          false, a, false, b, (short)0, acc[j], false, false);
    }
  }

  // --- epilogue: bias + RoPE, scatter to q/k staging buffers -------------
  const int t    = n0 >> 7;                    // head index (128 cols per head)
  const bool isK = (n0 & 127) >= 64;           // second half of head = k proj
  const int m0   = mtile * 16 + ((lane >> 4) << 3);
  const int ncb  = n0 + (lane & 15);
  _Float16* dst  = isK ? kbuf : qbuf;
#pragma unroll
  for (int j = 0; j < 4; ++j) {
    const int ncol = ncb + j * 16;
    const int d    = ncol & 63;
    const float bv = bias[ncol];
#pragma unroll
    for (int r = 0; r < 8; ++r) {
      const int mg = m0 + r;
      const int s  = mg & (SS - 1);
      const int b  = mg >> 8;
      float val = acc[j][r] + bv;
      // rotate-half partner lives in lane^1 (same m, d^1)
      float partner = __shfl_xor(val, 1, 32);
      float rot = (d & 1) ? partner : -partner;
      float res = val * costab[s * DD + d] + rot * sintab[s * DD + d];
      dst[((size_t)(b * TT + t) * SS + s) * DD + d] = (_Float16)res;
    }
  }
}

// ---------------------------------------------------------------------------
// Kernel 5: attention logits per (b,t): q(256x64) @ k(256x64)^T, K=64.
// Each wave: 16x64 strip, 2 unrolled K-steps. Epilogue: pad mask, causal
// tril(-1) penalty, * 1/sqrt(D).
// ---------------------------------------------------------------------------
__global__ __launch_bounds__(256) void gemm2_mask_kernel(
    const _Float16* __restrict__ qbuf,   // [B*T][S][D]
    const _Float16* __restrict__ kbuf,   // [B*T][S][D]
    const float*    __restrict__ mask,   // [B][S]
    float* __restrict__ out) {           // [B*T][S][S]
  const int lane  = threadIdx.x & 31;
  const int wid   = blockIdx.x * 8 + (threadIdx.x >> 5);  // 53248 waves total
  const int bt    = wid >> 6;                             // 832 (b,t) pairs
  const int rem   = wid & 63;
  const int mtile = rem >> 2;                             // 16 row tiles
  const int strip = rem & 3;                              // 4 strips of 64
  const int b     = bt / TT;

  const _Float16* qb = qbuf + (size_t)bt * SS * DD;
  const _Float16* kb = kbuf + (size_t)bt * SS * DD;

  v8f acc[4] = {};
#pragma unroll
  for (int k0 = 0; k0 < DD; k0 += 32) {
    v16h a = load_frag(qb + (size_t)mtile * 16 * DD + k0, DD, lane);
#pragma unroll
    for (int j = 0; j < 4; ++j) {
      v16h bf = load_frag(kb + (size_t)(strip * 64 + j * 16) * DD + k0, DD, lane);
      acc[j] = __builtin_amdgcn_wmma_f32_16x16x32_f16(
          false, a, false, bf, (short)0, acc[j], false, false);
    }
  }

  const int m0 = mtile * 16 + ((lane >> 4) << 3);
  const int nb = strip * 64 + (lane & 15);
  float* obase = out + (size_t)bt * SS * SS;
#pragma unroll
  for (int j = 0; j < 4; ++j) {
    const int n = nb + j * 16;
    const float pad = mask[b * SS + n];
    const float padterm = (1.0f - pad) * NEGC;
#pragma unroll
    for (int r = 0; r < 8; ++r) {
      const int m = m0 + r;
      float val = acc[j][r] * pad - padterm;
      if (m > n) val -= NEGC;                 // causal tril(ones,-1) penalty
      obase[(size_t)m * SS + n] = val * 0.125f;   // 1/sqrt(64)
    }
  }
}

// ---------------------------------------------------------------------------
// Host-side launch. Workspace layout (all 16B aligned):
//   Ah     f16 [M][H]        25,165,824 B
//   wT     f16 [N][H]         2,555,904 B
//   costab f32 [S][D]            65,536 B
//   sintab f32 [S][D]            65,536 B
//   qbuf   f16 [B*T][S][D]   27,262,976 B
//   kbuf   f16 [B*T][S][D]   27,262,976 B   (total ~82.4 MB)
// ---------------------------------------------------------------------------
extern "C" void kernel_launch(void* const* d_in, const int* in_sizes, int n_in,
                              void* d_out, int out_size, void* d_ws, size_t ws_size,
                              hipStream_t stream) {
  const float* hidden = (const float*)d_in[0];   // [B][S][H]
  const float* w      = (const float*)d_in[1];   // [H][N]
  const float* bias   = (const float*)d_in[2];   // [N]
  const float* mask   = (const float*)d_in[3];   // [B][S]
  float* out          = (float*)d_out;

  char* ws = (char*)d_ws;
  _Float16* Ah   = (_Float16*)ws;
  _Float16* wT   = (_Float16*)(ws + 25165824);
  float* costab  = (float*)(ws + 25165824 + 2555904);
  float* sintab  = costab + SS * DD;
  _Float16* qbuf = (_Float16*)((char*)(sintab + SS * DD));
  _Float16* kbuf = qbuf + (size_t)(BB * TT) * SS * DD;

  // 1) RoPE table
  rope_table_kernel<<<(SS * (DD / 2) + 255) / 256, 256, 0, stream>>>(costab, sintab);
  // 2) hidden f32 -> f16
  const int n4 = (MM * HH) / 4;                       // 3,145,728
  cvt_f16_kernel<<<(n4 + 255) / 256, 256, 0, stream>>>(hidden, Ah, n4);
  // 3) w -> wT f16
  transpose_w_kernel<<<(NN * HH + 255) / 256, 256, 0, stream>>>(w, wT);
  // 4) projection GEMM + RoPE  (26624 waves / 8 per block = 3328 blocks, exact)
  gemm1_rope_kernel<<<3328, 256, 0, stream>>>(Ah, wT, bias, costab, sintab,
                                              qbuf, kbuf);
  // 5) attention logits + mask (53248 waves / 8 per block = 6656 blocks, exact)
  gemm2_mask_kernel<<<6656, 256, 0, stream>>>(qbuf, kbuf, mask, out);
}
